// MotionPredictionLayer_33105607918032
// MI455X (gfx1250) — matrix operand order repaired and checked by reading, not compiled, not run on hardware
//
#include <hip/hip_runtime.h>
#include <hip/hip_bf16.h>

// Problem constants (from reference)
#define N_KNOTS 4
#define IN_DIM  512
#define OUT_DIM 512
#define BATCH   4096
#define KP      (N_KNOTS * IN_DIM)   // folded K dimension = 2048
#define KTILES  (KP / 32)            // 64 K-tiles of 32
#define NTILES  (OUT_DIM / 16)       // 32 N-tiles of 16
#define MTILES  (BATCH / 16)         // 256 M-tiles of 16

typedef __attribute__((ext_vector_type(16))) __bf16 v16bf;
typedef __attribute__((ext_vector_type(8)))  float  v8f;

__device__ __forceinline__ unsigned short f2bf(float f) {
    unsigned int u = __float_as_uint(f);
    u += 0x7FFFu + ((u >> 16) & 1u);   // round-to-nearest-even
    return (unsigned short)(u >> 16);
}

// ---------------------------------------------------------------------------
// Pack W[k][o][i] (fp32) -> Bpack bf16 tiles laid out exactly as the WMMA
// B-operand (32x16, 16-bit) wants per lane:
//   tile (kt, nt), lane l<16 : column n = nt*16+l, elements j<8 -> K = kt*32+j, j>=8 -> K = kt*32+8+j
//   lane l>=16             : column n = nt*16+(l-16), j<8 -> K = kt*32+8+j, j>=8 -> K = kt*32+16+j
// Layout: Bpack[((kt*NTILES + nt)*32 + lane)*16 + j]
// ---------------------------------------------------------------------------
__global__ void pack_w_kernel(const float* __restrict__ W,
                              unsigned int* __restrict__ Bp) {
    int t = blockIdx.x * blockDim.x + threadIdx.x;       // one dword (2 bf16)
    if (t >= KP * OUT_DIM / 2) return;
    int j2   = t & 7;
    int lane = (t >> 3) & 31;
    int nt   = (t >> 8) & (NTILES - 1);
    int kt   = t >> 13;
    int n    = nt * 16 + (lane & 15);
    unsigned int pack = 0;
#pragma unroll
    for (int e = 0; e < 2; ++e) {
        int j    = 2 * j2 + e;
        int kloc = (lane < 16) ? ((j < 8) ? j : 8 + j)
                               : ((j < 8) ? 8 + j : 16 + j);
        int kk   = kt * 32 + kloc;                       // folded K index
        int k    = kk >> 9;                              // knot
        int i    = kk & (IN_DIM - 1);                    // input feature
        float v  = W[((size_t)k * OUT_DIM + n) * IN_DIM + i];
        pack |= ((unsigned int)f2bf(v)) << (16 * e);
    }
    Bp[t] = pack;
}

// ---------------------------------------------------------------------------
// Build A'[b, k*512+i] = p[b,k]*feat[b,i] in bf16, pre-packed per WMMA
// A-operand layout (16-bit A 16x32):
//   lane l<16 : row m = mt*16+l, elements j<8 -> K=j, j>=8 -> K=8+j
//   lane l>=16: row m = mt*16+(l-16), j<8 -> K=8+j, j>=8 -> K=16+j
// Layout: Apack[((mt*KTILES + kt)*32 + lane)*16 + j]
// ---------------------------------------------------------------------------
__global__ void pack_a_kernel(const float* __restrict__ X,
                              unsigned int* __restrict__ Ap) {
    int t = blockIdx.x * blockDim.x + threadIdx.x;       // one dword (2 bf16)
    if (t >= BATCH * KP / 2) return;
    int j2   = t & 7;
    int lane = (t >> 3) & 31;
    int kt   = (t >> 8) & (KTILES - 1);
    int mt   = t >> 14;
    int b    = mt * 16 + (lane & 15);
    const float* xr = X + (size_t)b * (IN_DIM + N_KNOTS);
    unsigned int pack = 0;
#pragma unroll
    for (int e = 0; e < 2; ++e) {
        int j    = 2 * j2 + e;
        int kloc = (lane < 16) ? ((j < 8) ? j : 8 + j)
                               : ((j < 8) ? 8 + j : 16 + j);
        int kk   = kt * 32 + kloc;
        int k    = kk >> 9;
        int i    = kk & (IN_DIM - 1);
        float v  = xr[IN_DIM + k] * xr[i];               // p[b,k] * feat[b,i]
        pack |= ((unsigned int)f2bf(v)) << (16 * e);
    }
    Ap[t] = pack;
}

// ---------------------------------------------------------------------------
// GEMM: res = A'(4096x2048) * W'(2048x512) + p@bias, then ELU.
// 4 waves / block. Block tile M=64 x N=128. Each wave: M=64 x N=32 as a
// 4x2 grid of v_wmma_f32_16x16x32_bf16 tiles (6 fragment loads : 8 WMMAs).
// Grid (64, 4) = 256 blocks / 1024 waves.
// ---------------------------------------------------------------------------
union FragLd {
    uint4 q[2];
    v16bf v;
};

// byte strides in the packed layouts
#define A_KT_STRIDE 1024u             // 32 lanes * 32 B per (mt,kt) tile
#define A_MT_STRIDE (KTILES * 1024u)  // 64 KB per M-tile row
#define B_NT_STRIDE 1024u             // 32 lanes * 32 B per (kt,nt) tile
#define B_KT_STRIDE (NTILES * 1024u)  // 32 KB per K-tile row

__global__ __launch_bounds__(128)
void gemm_elu_kernel(const unsigned short* __restrict__ Ap,
                     const unsigned short* __restrict__ Bp,
                     const float* __restrict__ X,
                     const float* __restrict__ bias,
                     float* __restrict__ out) {
    const int lane = threadIdx.x & 31;
    const int wave = threadIdx.x >> 5;
    const int mtBase = blockIdx.x * 4;             // 4 M-tiles per block (all waves)
    const int ntBase = blockIdx.y * 8 + wave * 2;  // 2 N-tiles per wave

    v8f c[4][2];
#pragma unroll
    for (int mi = 0; mi < 4; ++mi)
#pragma unroll
        for (int ni = 0; ni < 2; ++ni) {
            v8f z = {0.f, 0.f, 0.f, 0.f, 0.f, 0.f, 0.f, 0.f};
            c[mi][ni] = z;
        }

    // Per-lane 32-bit byte offsets, computed once; K loop uses constant strides
    // that the backend can fold into the 24-bit instruction offset.
    const char* aBase = (const char*)Ap;
    const char* bBase = (const char*)Bp;
    unsigned aOff[4], bOff[2];
#pragma unroll
    for (int mi = 0; mi < 4; ++mi)
        aOff[mi] = (unsigned)(mtBase + mi) * A_MT_STRIDE + (unsigned)lane * 32u;
#pragma unroll
    for (int ni = 0; ni < 2; ++ni)
        bOff[ni] = (unsigned)(ntBase + ni) * B_NT_STRIDE + (unsigned)lane * 32u;

#pragma unroll 2
    for (int kt = 0; kt < KTILES; ++kt) {
        FragLd a[4], b[2];
#pragma unroll
        for (int ni = 0; ni < 2; ++ni) {
            const uint4* p =
                (const uint4*)(bBase + (bOff[ni] + (unsigned)kt * B_KT_STRIDE));
            b[ni].q[0] = p[0];
            b[ni].q[1] = p[1];
        }
#pragma unroll
        for (int mi = 0; mi < 4; ++mi) {
            const uint4* p =
                (const uint4*)(aBase + (aOff[mi] + (unsigned)kt * A_KT_STRIDE));
            a[mi].q[0] = p[0];
            a[mi].q[1] = p[1];
        }
#pragma unroll
        for (int mi = 0; mi < 4; ++mi)
#pragma unroll
            for (int ni = 0; ni < 2; ++ni) {
                // 8 args: (neg_a, A, neg_b, B, c_mod, C, reuse_a, reuse_b)
                c[mi][ni] = __builtin_amdgcn_wmma_f32_16x16x32_bf16(
                    false, a[mi].v, false, b[ni].v, (short)0, c[mi][ni],
                    false, false);
            }
    }

    // Epilogue: + p@bias (f32), ELU, store f32.
    // C layout: VGPR r, lanes 0-15 -> M = r, N = lane; lanes 16-31 -> M = 8+r.
    float bcol[2][N_KNOTS];
#pragma unroll
    for (int ni = 0; ni < 2; ++ni) {
        const int n = (ntBase + ni) * 16 + (lane & 15);
#pragma unroll
        for (int k = 0; k < N_KNOTS; ++k)
            bcol[ni][k] = bias[k * OUT_DIM + n];
    }

#pragma unroll
    for (int mi = 0; mi < 4; ++mi) {
        const int mrowBase = (mtBase + mi) * 16 + ((lane >> 4) << 3);
#pragma unroll
        for (int r = 0; r < 8; ++r) {
            const int m = mrowBase + r;
            const float* pk = X + (size_t)m * (IN_DIM + N_KNOTS) + IN_DIM;
            float p0 = pk[0], p1 = pk[1], p2 = pk[2], p3 = pk[3];
#pragma unroll
            for (int ni = 0; ni < 2; ++ni) {
                const int n = (ntBase + ni) * 16 + (lane & 15);
                float bs = p0 * bcol[ni][0] + p1 * bcol[ni][1] +
                           p2 * bcol[ni][2] + p3 * bcol[ni][3];
                float v = c[mi][ni][r] + bs;
                v = (v > 0.f) ? v : (__expf(v) - 1.f);   // ELU, alpha=1
                out[(size_t)m * OUT_DIM + n] = v;
            }
        }
    }
}

// ---------------------------------------------------------------------------
// Launch: workspace = Apack (16 MB bf16) + Bpack (2 MB bf16) = 18 MB.
// ---------------------------------------------------------------------------
extern "C" void kernel_launch(void* const* d_in, const int* in_sizes, int n_in,
                              void* d_out, int out_size, void* d_ws, size_t ws_size,
                              hipStream_t stream) {
    const float* x    = (const float*)d_in[0];   // (4096, 516)
    const float* W    = (const float*)d_in[1];   // (4, 512, 512)
    const float* bias = (const float*)d_in[2];   // (4, 512)
    float* out        = (float*)d_out;           // (4096, 512) fp32

    unsigned short* Ap = (unsigned short*)d_ws;              // 16 MB
    unsigned short* Bp = Ap + (size_t)BATCH * KP;            // + 2 MB

    {
        int total = KP * OUT_DIM / 2;
        pack_w_kernel<<<(total + 255) / 256, 256, 0, stream>>>(
            W, (unsigned int*)Bp);
    }
    {
        int total = BATCH * KP / 2;
        pack_a_kernel<<<(total + 255) / 256, 256, 0, stream>>>(
            x, (unsigned int*)Ap);
    }
    {
        dim3 grid(MTILES / 4, NTILES / 8);   // (64, 4)
        gemm_elu_kernel<<<grid, 128, 0, stream>>>(Ap, Bp, x, bias, out);
    }
}
